// CPUMoE_22995254902970
// MI455X (gfx1250) — compile-verified, matrix operand order checked by reading, not run on hardware
//
#include <hip/hip_runtime.h>

// ---------------------------------------------------------------------------
// MoE forward for MI455X (gfx1250, wave32, WMMA).
// Stages: rmsnorm+router -> per-expert [gate/up GEMM + SiLU*up] -> down GEMM.
// bf16 WMMA (V_WMMA_F32_16X16X32_BF16), fp32 accumulation, double-buffered LDS.
// Activation tiles move via TDM (TENSOR_LOAD_TO_LDS, one instr per tile) when
// available, else GLOBAL_LOAD_ASYNC_TO_LDS, else plain copies.
// ---------------------------------------------------------------------------

typedef __attribute__((ext_vector_type(16))) __bf16 v16bf;
typedef __attribute__((ext_vector_type(8)))  float  v8f;
typedef int v4i __attribute__((vector_size(16)));           // int4 pointee for async builtin
typedef unsigned v4u_t __attribute__((ext_vector_type(4))); // TDM D# group0
typedef int      v8i_t __attribute__((ext_vector_type(8))); // TDM D# group1 / arg5
typedef int      v4s_t __attribute__((ext_vector_type(4))); // TDM D# groups 2/3

#define H_DIM 2048
#define I_DIM 1408
#define N_EXP 8
#define MT 128   // token tile
#define NT 64    // output-col tile
#define KT 32    // K chunk per WMMA pass
#define XS 40    // LDS row stride (bf16 elems): 64B row + 16B pad = 80B, 16B-aligned slots

#if defined(__gfx1250__) && __has_builtin(__builtin_amdgcn_global_load_async_to_lds_b128) && __has_builtin(__builtin_amdgcn_s_wait_asynccnt)
#define HAS_ASYNC_LDS 1
#define WAIT_ASYNC(n) __builtin_amdgcn_s_wait_asynccnt(n)
#else
#define HAS_ASYNC_LDS 0
#define WAIT_ASYNC(n)
#endif

#if defined(__gfx1250__) && __has_builtin(__builtin_amdgcn_tensor_load_to_lds) && __has_builtin(__builtin_amdgcn_s_wait_tensorcnt)
#define HAS_TDM 1
#else
#define HAS_TDM 0
#endif

// 16-byte global -> LDS copy; async (ASYNCcnt-tracked, no VGPR staging) when available.
__device__ inline void cp16_g2l(const __bf16* gsrc, __bf16* ldst) {
#if HAS_ASYNC_LDS
    __builtin_amdgcn_global_load_async_to_lds_b128(
        (__attribute__((address_space(1))) v4i*)(unsigned long long)(const void*)gsrc,
        (__attribute__((address_space(3))) v4i*)(unsigned)(unsigned long long)(void*)ldst,
        0, 0);
#else
    *(uint4*)ldst = *(const uint4*)gsrc;
#endif
}

#if HAS_TDM
// One-instruction DMA of a [rows x KT] bf16 tile (row-major, leading dim `ld`
// elements) into LDS with the XS-stride layout produced by TDM padding:
// 16 DWORDs (64B row) + 4 DWORDs pad = 80B = XS*2 bytes.
__device__ inline void tdm_load_tile(const __bf16* gsrc, __bf16* ldst,
                                     unsigned rows, unsigned ld) {
    const unsigned long long ga = (unsigned long long)(const void*)gsrc;
    const unsigned la = (unsigned)(unsigned long long)(void*)ldst;
    v4u_t g0;
    g0[0] = 1u;                                   // count=1 valid user descriptor
    g0[1] = la;                                   // lds_addr (bytes)
    g0[2] = (unsigned)ga;                         // global_addr[31:0]
    g0[3] = (unsigned)((ga >> 32) & 0x1FFFFFFull) // global_addr[56:32]
            | (2u << 30);                         // type = 2 ("image")
    v8i_t g1;
    g1[0] = (int)((1u << 16)    // data_size = 1 -> 2 bytes
                | (1u << 20)    // pad_enable
                | (3u << 22)    // pad_interval: every 16 DWORDs (64B)
                | (3u << 25));  // pad_amount: 4 DWORDs (16B)
    g1[1] = (int)((ld & 0xFFFFu) << 16);                        // tensor_dim0[15:0]
    g1[2] = (int)(((ld >> 16) & 0xFFFFu) | ((rows & 0xFFFFu) << 16)); // dim0 hi | dim1 lo
    g1[3] = (int)(((rows >> 16) & 0xFFFFu) | ((unsigned)KT << 16));   // dim1 hi | tile_dim0
    g1[4] = (int)(rows & 0xFFFFu);                              // tile_dim1 (tile_dim2 = 0)
    g1[5] = (int)ld;                                            // tensor_dim0_stride[31:0]
    g1[6] = 0;                                                  // stride0 hi | stride1 lo
    g1[7] = 0;                                                  // stride1 hi
    const v4s_t gz4 = {0, 0, 0, 0};                             // groups 2/3 unused (2D)
    const v8i_t gz8 = {0, 0, 0, 0, 0, 0, 0, 0};                 // trailing group (unused)
    __builtin_amdgcn_tensor_load_to_lds(g0, g1, gz4, gz4, gz8, 0);
}
#endif

// Stage a [MT x KT] bf16 activation tile into LDS (TDM: wave 0 issues 1 op).
__device__ inline void stage_act(const __bf16* g, int ld, int row0, int k0,
                                 __bf16* l, int tid) {
#if HAS_TDM
    if (tid < 32) tdm_load_tile(g + (size_t)row0 * ld + k0, l, MT, (unsigned)ld);
#else
    const int r = tid >> 1, c = (tid & 1) * 16;
    const __bf16* gp = g + (size_t)(row0 + r) * ld + k0 + c;
    __bf16* sp = l + r * XS + c;
    cp16_g2l(gp, sp);
    cp16_g2l(gp + 8, sp + 8);
#endif
}

// Wait for the previous stage's activation-tile DMA to land in LDS.
__device__ inline void wait_stage_pipelined(int tid) {
#if HAS_TDM
    if (tid < 32) __builtin_amdgcn_s_wait_tensorcnt(1);
#else
    WAIT_ASYNC(2);
#endif
}
__device__ inline void wait_stage_last(int tid) {
#if HAS_TDM
    if (tid < 32) __builtin_amdgcn_s_wait_tensorcnt(0);
#else
    WAIT_ASYNC(0);
#endif
}

// Load a 16x32 bf16 fragment (A-layout; B uses same layout because we stage the
// weight tile as [N][K] in LDS). Per ISA: lane holds row (lane&15); lanes 0-15
// hold K 0..7 & 16..23, lanes 16-31 hold K 8..15 & 24..31.
__device__ inline v16bf load_frag(const __bf16* lds, int rowBase, int lane) {
    const __bf16* p = lds + (rowBase + (lane & 15)) * XS + ((lane >> 4) << 3);
    v16bf f;
#pragma unroll
    for (int j = 0; j < 8; ++j) { f[j] = p[j]; f[j + 8] = p[j + 16]; }
    return f;
}

// SiLU with fast reciprocal (v_rcp_f32) instead of IEEE division.
__device__ inline float silu(float g) {
    return g * __builtin_amdgcn_rcpf(1.0f + __expf(-g));
}

// Stage a [NT x KT] weight tile as [N][K] with fp32 -> bf16 convert.
__device__ inline void stage_w(const float* w, int ld, int n0, int k0,
                               __bf16* l, int tid) {
    const int r = tid >> 2, c = (tid & 3) * 8;
    const float* gp = w + (size_t)(n0 + r) * ld + k0 + c;
    __builtin_prefetch(gp + 2 * KT, 0, 1);  // global_prefetch_b8 two chunks ahead
    __bf16* sp = l + r * XS + c;
#pragma unroll
    for (int j = 0; j < 8; ++j) sp[j] = (__bf16)gp[j];
}

// ---------------------------------------------------------------------------
// Kernel 1: RMSNorm + router logits + softmax + top-2 -> combine weights.
// One block per token, 256 threads, 8 contiguous elements per thread.
// ---------------------------------------------------------------------------
__global__ __launch_bounds__(256)
void moe_rms_router(const float* __restrict__ x,
                    const float* __restrict__ rmsw,
                    const float* __restrict__ routerw,
                    __bf16* __restrict__ xb,
                    float* __restrict__ cw,
                    float* __restrict__ out,
                    float* __restrict__ logits) {
    __shared__ float red[256];
    __shared__ float pr[N_EXP * 256];
    const int t   = blockIdx.x;
    const int tid = threadIdx.x;
    const float* xr = x + (size_t)t * H_DIM + tid * 8;

    float xv[8];
    float ss = 0.f;
#pragma unroll
    for (int j = 0; j < 8; ++j) { xv[j] = xr[j]; ss += xv[j] * xv[j]; }
    red[tid] = ss;
    __syncthreads();
    for (int s = 128; s > 0; s >>= 1) {
        if (tid < s) red[tid] += red[tid + s];
        __syncthreads();
    }
    const float scale = rsqrtf(red[0] * (1.0f / H_DIM) + 1e-6f);

    float part[N_EXP];
#pragma unroll
    for (int e = 0; e < N_EXP; ++e) part[e] = 0.f;

    const int base = tid * 8;
#pragma unroll
    for (int j = 0; j < 8; ++j) {
        const float xn = xv[j] * scale * rmsw[base + j];
        xb[(size_t)t * H_DIM + base + j]  = (__bf16)xn;
        out[(size_t)t * H_DIM + base + j] = 0.f;   // init output accumulator
#pragma unroll
        for (int e = 0; e < N_EXP; ++e)
            part[e] += xn * routerw[e * H_DIM + base + j];
    }
#pragma unroll
    for (int e = 0; e < N_EXP; ++e) pr[e * 256 + tid] = part[e];
    __syncthreads();
    for (int s = 128; s > 0; s >>= 1) {
        if (tid < s)
#pragma unroll
            for (int e = 0; e < N_EXP; ++e) pr[e * 256 + tid] += pr[e * 256 + tid + s];
        __syncthreads();
    }
    if (tid == 0) {
        float l[N_EXP], p[N_EXP];
        float m = -3.4e38f;
#pragma unroll
        for (int e = 0; e < N_EXP; ++e) { l[e] = pr[e * 256]; m = fmaxf(m, l[e]); }
        float s = 0.f;
#pragma unroll
        for (int e = 0; e < N_EXP; ++e) { p[e] = __expf(l[e] - m); s += p[e]; }
        const float inv_s = 1.f / s;
        int i1 = 0;
#pragma unroll
        for (int e = 1; e < N_EXP; ++e) if (p[e] > p[i1]) i1 = e;
        int i2 = (i1 == 0) ? 1 : 0;
#pragma unroll
        for (int e = 0; e < N_EXP; ++e) if (e != i1 && p[e] > p[i2]) i2 = e;
        const float w1 = p[i1] * inv_s, w2 = p[i2] * inv_s;
        const float inv = 1.f / (w1 + w2 + 1e-20f);
#pragma unroll
        for (int e = 0; e < N_EXP; ++e) {
            logits[(size_t)t * N_EXP + e] = l[e];
            cw[(size_t)t * N_EXP + e] = 0.f;
        }
        cw[(size_t)t * N_EXP + i1] = w1 * inv;
        cw[(size_t)t * N_EXP + i2] = w2 * inv;
    }
}

// ---------------------------------------------------------------------------
// Kernel 2 (per expert): h = silu(x @ wg^T) * (x @ wu^T), bf16 out.
// Block tile 128x64; 8 waves, each owns a 32x32 quadrant (2x2 WMMA tiles).
// Double-buffered LDS; x tile via TDM/async-to-LDS, weight tiles cvt-staged.
// ---------------------------------------------------------------------------
__global__ __launch_bounds__(256)
void moe_gateup(const __bf16* __restrict__ xb,
                const float* __restrict__ wg,
                const float* __restrict__ wu,
                __bf16* __restrict__ hbuf) {
    __shared__ __bf16 lx[2][MT * XS];
    __shared__ __bf16 lg[2][NT * XS];
    __shared__ __bf16 lu[2][NT * XS];
    const int tid  = threadIdx.x;
    const int lane = tid & 31;
    const int wv   = tid >> 5;
    const int wm   = (wv & 3) * 32;
    const int wn   = (wv >> 2) * 32;
    const int bm   = blockIdx.x * MT;
    const int bn   = blockIdx.y * NT;

    v8f ag[4], au[4];
    const v8f zero = {};
#pragma unroll
    for (int i = 0; i < 4; ++i) { ag[i] = zero; au[i] = zero; }

    // prologue: stage chunk 0 into buffer 0
    stage_act(xb, H_DIM, bm, 0, lx[0], tid);
    stage_w(wg, H_DIM, bn, 0, lg[0], tid);
    stage_w(wu, H_DIM, bn, 0, lu[0], tid);

    for (int k0 = 0; k0 < H_DIM; k0 += KT) {
        const int cur = (k0 / KT) & 1;
        const int nxt = cur ^ 1;
        const bool has_next = (k0 + KT) < H_DIM;
        if (has_next) {
            stage_act(xb, H_DIM, bm, k0 + KT, lx[nxt], tid);
            stage_w(wg, H_DIM, bn, k0 + KT, lg[nxt], tid);
            stage_w(wu, H_DIM, bn, k0 + KT, lu[nxt], tid);
            wait_stage_pipelined(tid);   // cur-stage tile DMA has landed
        } else {
            wait_stage_last(tid);
        }
        __syncthreads();  // cur buffer fully populated by all waves
        const v16bf a0  = load_frag(lx[cur], wm + 0,  lane);
        const v16bf a1  = load_frag(lx[cur], wm + 16, lane);
        const v16bf bg0 = load_frag(lg[cur], wn + 0,  lane);
        const v16bf bg1 = load_frag(lg[cur], wn + 16, lane);
        const v16bf bu0 = load_frag(lu[cur], wn + 0,  lane);
        const v16bf bu1 = load_frag(lu[cur], wn + 16, lane);
        ag[0] = __builtin_amdgcn_wmma_f32_16x16x32_bf16(false, a0, false, bg0, (short)0, ag[0], false, false);
        ag[1] = __builtin_amdgcn_wmma_f32_16x16x32_bf16(false, a0, false, bg1, (short)0, ag[1], false, false);
        ag[2] = __builtin_amdgcn_wmma_f32_16x16x32_bf16(false, a1, false, bg0, (short)0, ag[2], false, false);
        ag[3] = __builtin_amdgcn_wmma_f32_16x16x32_bf16(false, a1, false, bg1, (short)0, ag[3], false, false);
        au[0] = __builtin_amdgcn_wmma_f32_16x16x32_bf16(false, a0, false, bu0, (short)0, au[0], false, false);
        au[1] = __builtin_amdgcn_wmma_f32_16x16x32_bf16(false, a0, false, bu1, (short)0, au[1], false, false);
        au[2] = __builtin_amdgcn_wmma_f32_16x16x32_bf16(false, a1, false, bu0, (short)0, au[2], false, false);
        au[3] = __builtin_amdgcn_wmma_f32_16x16x32_bf16(false, a1, false, bu1, (short)0, au[3], false, false);
        __syncthreads();  // reads of cur done before it is overwritten next round
    }

    // epilogue: SiLU-GLU, store bf16. C layout: M = (lane>>4)*8 + r, N = lane&15.
    const int mh = (lane >> 4) * 8, nn = lane & 15;
#pragma unroll
    for (int ti = 0; ti < 4; ++ti) {
        const int mi = (ti >> 1) * 16, ni = (ti & 1) * 16;
#pragma unroll
        for (int r = 0; r < 8; ++r) {
            const int row = bm + wm + mi + mh + r;
            const int col = bn + wn + ni + nn;
            const float h = silu(ag[ti][r]) * au[ti][r];
            hbuf[(size_t)row * I_DIM + col] = (__bf16)h;
        }
    }
}

// ---------------------------------------------------------------------------
// Kernel 3 (per expert): out += cw[:,e] * (h @ wd^T). Same tiling, K over I.
// ---------------------------------------------------------------------------
__global__ __launch_bounds__(256)
void moe_down(const __bf16* __restrict__ hbuf,
              const float* __restrict__ wd,
              const float* __restrict__ cw,
              float* __restrict__ out, int e) {
    __shared__ __bf16 lh[2][MT * XS];
    __shared__ __bf16 lw[2][NT * XS];
    const int tid  = threadIdx.x;
    const int lane = tid & 31;
    const int wv   = tid >> 5;
    const int wm   = (wv & 3) * 32;
    const int wn   = (wv >> 2) * 32;
    const int bm   = blockIdx.x * MT;
    const int bn   = blockIdx.y * NT;

    v8f acc[4];
    const v8f zero = {};
#pragma unroll
    for (int i = 0; i < 4; ++i) acc[i] = zero;

    stage_act(hbuf, I_DIM, bm, 0, lh[0], tid);
    stage_w(wd, I_DIM, bn, 0, lw[0], tid);

    for (int k0 = 0; k0 < I_DIM; k0 += KT) {
        const int cur = (k0 / KT) & 1;
        const int nxt = cur ^ 1;
        const bool has_next = (k0 + KT) < I_DIM;
        if (has_next) {
            stage_act(hbuf, I_DIM, bm, k0 + KT, lh[nxt], tid);
            stage_w(wd, I_DIM, bn, k0 + KT, lw[nxt], tid);
            wait_stage_pipelined(tid);
        } else {
            wait_stage_last(tid);
        }
        __syncthreads();
        const v16bf a0 = load_frag(lh[cur], wm + 0,  lane);
        const v16bf a1 = load_frag(lh[cur], wm + 16, lane);
        const v16bf b0 = load_frag(lw[cur], wn + 0,  lane);
        const v16bf b1 = load_frag(lw[cur], wn + 16, lane);
        acc[0] = __builtin_amdgcn_wmma_f32_16x16x32_bf16(false, a0, false, b0, (short)0, acc[0], false, false);
        acc[1] = __builtin_amdgcn_wmma_f32_16x16x32_bf16(false, a0, false, b1, (short)0, acc[1], false, false);
        acc[2] = __builtin_amdgcn_wmma_f32_16x16x32_bf16(false, a1, false, b0, (short)0, acc[2], false, false);
        acc[3] = __builtin_amdgcn_wmma_f32_16x16x32_bf16(false, a1, false, b1, (short)0, acc[3], false, false);
        __syncthreads();
    }

    // weighted accumulate into out (experts are serialized on the stream)
    const int mh = (lane >> 4) * 8, nn = lane & 15;
#pragma unroll
    for (int ti = 0; ti < 4; ++ti) {
        const int mi = (ti >> 1) * 16, ni = (ti & 1) * 16;
#pragma unroll
        for (int r = 0; r < 8; ++r) {
            const int row = bm + wm + mi + mh + r;
            const int col = bn + wn + ni + nn;
            const float cwv = cw[(size_t)row * N_EXP + e];
            out[(size_t)row * H_DIM + col] += cwv * acc[ti][r];
        }
    }
}

// ---------------------------------------------------------------------------
extern "C" void kernel_launch(void* const* d_in, const int* in_sizes, int n_in,
                              void* d_out, int out_size, void* d_ws, size_t ws_size,
                              hipStream_t stream) {
    const float* x        = (const float*)d_in[0];
    const float* rmsw     = (const float*)d_in[1];
    const float* routerw  = (const float*)d_in[2];
    const float* wg_all   = (const float*)d_in[3];
    const float* wu_all   = (const float*)d_in[4];
    const float* wd_all   = (const float*)d_in[5];

    const int T = in_sizes[0] / H_DIM;  // 8192 tokens

    float* out    = (float*)d_out;
    float* logits = out + (size_t)T * H_DIM;

    // workspace layout: xb (bf16 T*H) | cw (f32 T*E) | h (bf16 T*I, reused per expert)
    char* ws = (char*)d_ws;
    __bf16* xb  = (__bf16*)ws;
    float*  cw  = (float*)(ws + (size_t)T * H_DIM * 2);
    __bf16* hbf = (__bf16*)(ws + (size_t)T * H_DIM * 2 + (size_t)T * N_EXP * 4);

    moe_rms_router<<<T, 256, 0, stream>>>(x, rmsw, routerw, xb, cw, out, logits);

    const dim3 g2(T / MT, I_DIM / NT);  // 64 x 22
    const dim3 g3(T / MT, H_DIM / NT);  // 64 x 32
    for (int e = 0; e < N_EXP; ++e) {
        // one expert per launch pair -> expert's weights (23 MB fp32) stay L2-resident
        moe_gateup<<<g2, 256, 0, stream>>>(xb,
                                           wg_all + (size_t)e * I_DIM * H_DIM,
                                           wu_all + (size_t)e * I_DIM * H_DIM,
                                           hbf);
        moe_down<<<g3, 256, 0, stream>>>(hbf,
                                         wd_all + (size_t)e * H_DIM * I_DIM,
                                         cw, out, e);
    }
}
